// R_GAMLP_54924041781389
// MI455X (gfx1250) — matrix operand order repaired and checked by reading, not compiled, not run on hardware
//
#include <hip/hip_runtime.h>

#define Hh   8
#define Nn   100000
#define Ff   256
#define HIDD 512
#define Cc   100
#define CPAD 112
#define TILE_D0 (16 * Ff)   // 4096 elements per TDM line (16 nodes x 256 f)

typedef __attribute__((ext_vector_type(16))) __bf16 v16bf;
typedef __attribute__((ext_vector_type(8)))  __bf16 v8bf;
typedef __attribute__((ext_vector_type(8)))  float  v8f;
typedef __attribute__((ext_vector_type(4)))  unsigned int v4u;
typedef __attribute__((ext_vector_type(8)))  int    v8i;
typedef __attribute__((ext_vector_type(4)))  int    v4i;

__device__ __forceinline__ __bf16 f2bf(float f) {
  unsigned u = __builtin_bit_cast(unsigned, f);
  unsigned r = u + 0x7FFFu + ((u >> 16) & 1u);   // round-to-nearest-even
  unsigned short s = (unsigned short)(r >> 16);
  return __builtin_bit_cast(__bf16, s);
}

// A-matrix 16x32 bf16 fragment (ISA 7.12.2): lane<16 holds M=lane,
// elems 0..7 -> K=k0+half*8.., elems 8..15 -> K=k0+16+half*8..
__device__ __forceinline__ v16bf load_fragA(const __bf16* rowbase, int k0, int half) {
  const v8bf* p0 = reinterpret_cast<const v8bf*>(rowbase + k0 + half * 8);
  const v8bf* p1 = reinterpret_cast<const v8bf*>(rowbase + k0 + 16 + half * 8);
  v8bf a0 = *p0, a1 = *p1;
  v16bf r;
  #pragma unroll
  for (int i = 0; i < 8; ++i) { r[i] = a0[i]; r[i + 8] = a1[i]; }
  return r;
}

// B-matrix 32x16 bf16 fragment: lane holds column n=lane%16,
// elems 0..7 -> K=k0+half*8.., elems 8..15 -> K=k0+16+half*8..
__device__ __forceinline__ v16bf load_fragB(const __bf16* colbase, int k0, int half) {
  const v8bf* p0 = reinterpret_cast<const v8bf*>(colbase + k0 + half * 8);
  const v8bf* p1 = reinterpret_cast<const v8bf*>(colbase + k0 + 16 + half * 8);
  v8bf a0 = *p0, a1 = *p1;
  v16bf r;
  #pragma unroll
  for (int i = 0; i < 8; ++i) { r[i] = a0[i]; r[i + 8] = a1[i]; }
  return r;
}

__global__ void prep_weights(const float* __restrict__ fc1_w,
                             const float* __restrict__ fc2_w,
                             __bf16* __restrict__ w1b,
                             __bf16* __restrict__ w2p) {
  int i = blockIdx.x * blockDim.x + threadIdx.x;
  if (i < HIDD * Ff) w1b[i] = f2bf(fc1_w[i]);
  if (i < CPAD * HIDD) {
    int row = i / HIDD, k = i % HIDD;
    float v = (row < Cc) ? fc2_w[row * HIDD + k] : 0.0f;
    w2p[i] = f2bf(v);
  }
}

__global__ __launch_bounds__(256)
void gamlp_fused(const float* __restrict__ features,
                 const float* __restrict__ lr_att_w,
                 const float* __restrict__ lr_att_b,
                 const float* __restrict__ fc1_b,
                 const float* __restrict__ fc2_b,
                 const float* __restrict__ prelu_w,
                 const __bf16* __restrict__ w1b,
                 const __bf16* __restrict__ w2p,
                 float* __restrict__ out) {
  __shared__ alignas(16) float feat_lds[Hh * 16 * Ff];  // 128 KB: [hop][node][f]
  __shared__ float s_hp[16 * Hh];
  __shared__ float s_cp[16 * Hh];
  __shared__ float s_att[16 * Hh];
  __shared__ alignas(16) __bf16 right_lds[16 * Ff];     // 8 KB
  __shared__ alignas(16) __bf16 h_lds[16 * HIDD];       // 16 KB

  const int tid  = threadIdx.x;
  const int wave = tid >> 5;
  const int lane = tid & 31;
  const int node0 = blockIdx.x * 16;

  // ---------- TDM: async-stage the 8x(16x256) fp32 feature tile into LDS ----------
  if (wave == 0) {
    unsigned long long ga =
        (unsigned long long)(uintptr_t)features + (unsigned long long)node0 * (Ff * 4);
    unsigned ga_lo = (unsigned)ga;
    unsigned ga_hi = (unsigned)(ga >> 32);
    unsigned lds_off = (unsigned)(uintptr_t)(&feat_lds[0]);

    v4u g0;
    g0[0] = 1u;                                   // count=1 (valid), user mode
    g0[1] = lds_off;                              // lds_addr (bytes)
    g0[2] = ga_lo;                                // global_addr[31:0]
    g0[3] = (ga_hi & 0x01FFFFFFu) | (2u << 30);   // global_addr[56:32] | type=2

    v8i g1;
    g1[0] = (int)(2u << 16);                            // data_size=2 -> 4 bytes
    g1[1] = (int)((unsigned)(TILE_D0 & 0xFFFF) << 16);  // tensor_dim0 lo16
    g1[2] = (int)(((unsigned)TILE_D0 >> 16) | (8u << 16)); // dim0 hi | tensor_dim1 lo
    g1[3] = (int)((unsigned)TILE_D0 << 16);             // dim1 hi (0) | tile_dim0
    g1[4] = 8;                                          // tile_dim1=8, tile_dim2=0
    g1[5] = (int)(Nn * Ff);                             // tensor_dim0_stride lo32
    g1[6] = 0;                                          // stride0 hi | stride1 lo
    g1[7] = 0;                                          // stride1 hi

    v4i gz4;
    gz4[0] = 0; gz4[1] = 0; gz4[2] = 0; gz4[3] = 0;
    v8i gz8;
    #pragma unroll
    for (int i = 0; i < 8; ++i) gz8[i] = 0;

    __builtin_amdgcn_tensor_load_to_lds(g0, g1, gz4, gz4, gz8, 0);
    __builtin_amdgcn_s_wait_tensorcnt(0);
  }

  // Overlap with DMA: preload attention weight slices from global.
  float whi[8], wcu[8];
  #pragma unroll
  for (int j = 0; j < 8; ++j) {
    whi[j] = lr_att_w[lane * 8 + j];
    wcu[j] = lr_att_w[Ff + lane * 8 + j];
  }
  __syncthreads();   // publish TDM-written LDS to all waves

  // ---------- Phase A: hist/cur projections (one hop per wave), LDS reads ----------
  {
    const int hop = wave;
    for (int n = 0; n < 16; ++n) {
      const float* row = feat_lds + (hop * 16 + n) * Ff + lane * 8;
      float4 x0 = *reinterpret_cast<const float4*>(row);
      float4 x1 = *reinterpret_cast<const float4*>(row + 4);
      float xv[8] = {x0.x, x0.y, x0.z, x0.w, x1.x, x1.y, x1.z, x1.w};
      float dh = 0.f, dc = 0.f;
      #pragma unroll
      for (int j = 0; j < 8; ++j) { dh += xv[j] * whi[j]; dc += xv[j] * wcu[j]; }
      #pragma unroll
      for (int off = 16; off > 0; off >>= 1) {
        dh += __shfl_xor(dh, off, 32);
        dc += __shfl_xor(dc, off, 32);
      }
      if (lane == 0) { s_hp[n * Hh + hop] = dh; s_cp[n * Hh + hop] = dc; }
    }
  }
  __syncthreads();

  // ---------- Phase B: scalar attention recursion (1 thread / node) ----------
  if (tid < 16) {
    const float b = lr_att_b[0];
    float hp[Hh], cp[Hh], sc[Hh];
    #pragma unroll
    for (int h = 0; h < Hh; ++h) { hp[h] = s_hp[tid * Hh + h]; cp[h] = s_cp[tid * Hh + h]; }
    sc[0] = fmaxf(hp[0] + cp[0] + b, 0.f);
    for (int i = 1; i < Hh; ++i) {
      float m = sc[0];
      for (int j = 1; j < i; ++j) m = fmaxf(m, sc[j]);
      float den = 0.f, num = 0.f;
      for (int j = 0; j < i; ++j) { float e = __expf(sc[j] - m); den += e; num += e * hp[j]; }
      sc[i] = fmaxf(num / den + cp[i] + b, 0.f);
    }
    float m = sc[0];
    for (int j = 1; j < Hh; ++j) m = fmaxf(m, sc[j]);
    float e[Hh], den = 0.f;
    for (int j = 0; j < Hh; ++j) { e[j] = __expf(sc[j] - m); den += e[j]; }
    float inv = 1.f / den;
    for (int j = 0; j < Hh; ++j) s_att[tid * Hh + j] = e[j] * inv;
  }
  __syncthreads();

  // ---------- Phase C: right = att-weighted hop combo (LDS->LDS, bf16) ----------
  for (int n = 0; n < 16; ++n) {
    float aw[Hh];
    #pragma unroll
    for (int h = 0; h < Hh; ++h) aw[h] = s_att[n * Hh + h];
    float acc = 0.f;
    #pragma unroll
    for (int h = 0; h < Hh; ++h)
      acc += aw[h] * feat_lds[(h * 16 + n) * Ff + tid];
    right_lds[n * Ff + tid] = f2bf(acc);
  }
  __syncthreads();

  const int half = lane >> 4;
  const int nn   = lane & 15;

  // ---------- Phase D: GEMM1 (16x256 @ 256x512) + PReLU, WMMA bf16 ----------
  {
    const float pw = prelu_w[0];
    v8f acc[4];
    #pragma unroll
    for (int t = 0; t < 4; ++t)
      #pragma unroll
      for (int i = 0; i < 8; ++i) acc[t][i] = 0.f;

    const __bf16* arow = right_lds + nn * Ff;
    #pragma unroll
    for (int kk = 0; kk < 8; ++kk) {
      v16bf af = load_fragA(arow, kk * 32, half);
      #pragma unroll
      for (int t = 0; t < 4; ++t) {
        const int col = wave * 64 + t * 16 + nn;
        v16bf bf = load_fragB(w1b + (size_t)col * Ff, kk * 32, half);
        acc[t] = __builtin_amdgcn_wmma_f32_16x16x32_bf16(
            false, af, false, bf, (short)0, acc[t], false, false);
      }
    }
    #pragma unroll
    for (int t = 0; t < 4; ++t) {
      const int col = wave * 64 + t * 16 + nn;
      const float bias = fc1_b[col];
      #pragma unroll
      for (int r = 0; r < 8; ++r) {
        const int m = r + half * 8;
        float v = acc[t][r] + bias;
        v = (v >= 0.f) ? v : pw * v;
        h_lds[m * HIDD + col] = f2bf(v);
      }
    }
  }
  __syncthreads();

  // ---------- Phase E: GEMM2 (16x512 @ 512x112pad), bias, fp32 out ----------
  if (wave < 7) {
    v8f acc;
    #pragma unroll
    for (int i = 0; i < 8; ++i) acc[i] = 0.f;

    const __bf16* arow = h_lds + nn * HIDD;
    const int col = wave * 16 + nn;
    const __bf16* bcol = w2p + (size_t)col * HIDD;
    #pragma unroll
    for (int kk = 0; kk < 16; ++kk) {
      v16bf af = load_fragA(arow, kk * 32, half);
      v16bf bf = load_fragB(bcol, kk * 32, half);
      acc = __builtin_amdgcn_wmma_f32_16x16x32_bf16(
          false, af, false, bf, (short)0, acc, false, false);
    }
    if (col < Cc) {
      const float bias = fc2_b[col];
      #pragma unroll
      for (int r = 0; r < 8; ++r) {
        const int m = r + half * 8;
        out[(size_t)(node0 + m) * Cc + col] = acc[r] + bias;
      }
    }
  }
}

extern "C" void kernel_launch(void* const* d_in, const int* in_sizes, int n_in,
                              void* d_out, int out_size, void* d_ws, size_t ws_size,
                              hipStream_t stream) {
  (void)in_sizes; (void)n_in; (void)out_size; (void)ws_size;
  const float* features = (const float*)d_in[0];
  const float* lr_att_w = (const float*)d_in[1];
  const float* lr_att_b = (const float*)d_in[2];
  const float* fc1_w    = (const float*)d_in[3];
  const float* fc1_b    = (const float*)d_in[4];
  const float* fc2_w    = (const float*)d_in[5];
  const float* fc2_b    = (const float*)d_in[6];
  const float* prelu_w  = (const float*)d_in[7];

  __bf16* w1b = (__bf16*)d_ws;                     // 512*256 bf16
  __bf16* w2p = w1b + (size_t)HIDD * Ff;           // 112*512 bf16 (zero-padded)

  prep_weights<<<(HIDD * Ff + 255) / 256, 256, 0, stream>>>(fc1_w, fc2_w, w1b, w2p);
  gamlp_fused<<<Nn / 16, 256, 0, stream>>>(features, lr_att_w, lr_att_b,
                                           fc1_b, fc2_b, prelu_w,
                                           w1b, w2p, (float*)d_out);
}
